// LongformerEncoder_30872224923761
// MI455X (gfx1250) — compile-verified, matrix operand order checked by reading, not run on hardware
//
#include <hip/hip_runtime.h>
#include <hip/hip_bf16.h>

typedef __attribute__((ext_vector_type(16))) _Float16 v16h;
typedef __attribute__((ext_vector_type(8)))  float    v8f;
typedef __attribute__((ext_vector_type(4)))  unsigned int u32x4;
typedef __attribute__((ext_vector_type(8)))  int      i32x8;
typedef __attribute__((ext_vector_type(4)))  int      i32x4;

#define B_    2
#define S_    4096
#define D_    768
#define H_    12
#define L_    4
#define W_    128
#define DFF_  3072
#define DH_   64
#define NC_   (S_ / W_)     // 32 chunks
#define MROWS (B_ * S_)     // 8192 token rows

union FragH {
    v16h v;
    uint4 q[2];
    _Float16 h[16];
};

__device__ inline v8f zero8() {
    v8f z = {0.f, 0.f, 0.f, 0.f, 0.f, 0.f, 0.f, 0.f};
    return z;
}

__device__ inline v8f wmma16(v16h a, v16h b, v8f c) {
    // D = A(16xK f16) * B(Kx16 f16) + C(16x16 f32), K=32
    return __builtin_amdgcn_wmma_f32_16x16x32_f16(
        false, a, false, b, (short)0, c, false, false);
}

// convert 8 consecutive fp32 -> f16 with scale
__device__ inline void load8_h(_Float16* dst, const float* p, float scale) {
    float4 a = *(const float4*)p;
    float4 b = *(const float4*)(p + 4);
    dst[0] = (_Float16)(a.x * scale); dst[1] = (_Float16)(a.y * scale);
    dst[2] = (_Float16)(a.z * scale); dst[3] = (_Float16)(a.w * scale);
    dst[4] = (_Float16)(b.x * scale); dst[5] = (_Float16)(b.y * scale);
    dst[6] = (_Float16)(b.z * scale); dst[7] = (_Float16)(b.w * scale);
}

// ---------------------------------------------------------------------------
// GEMM: C[M,N] = act(A[M,K] @ Bw[K,N] + bias[N]); tiles 128x64, BK=32, 8 waves
// A tile staged via Tensor Data Mover (async DMA global->LDS with row padding)
// ---------------------------------------------------------------------------
__global__ __launch_bounds__(256)
void gemm_kernel(const float* __restrict__ A, const float* __restrict__ Bw,
                 const float* __restrict__ bias, float* __restrict__ C,
                 int M, int K, int N, int relu)
{
    // A tile: 128 rows x 32 fp32, TDM pads each 32-DWORD row with 8 DWORDs
    // -> effective row stride 40 floats (160B, 16B-aligned groups)
    __shared__ float    As32[128 * 40];
    __shared__ _Float16 Bs[64 * 40];    // [n][k] (transposed f16), stride 40

    const int tid  = threadIdx.x;
    const int lane = tid & 31;
    const int wv   = tid >> 5;          // 8 waves
    const int wm   = wv >> 1;           // 0..3 : 32-row strip
    const int wn   = wv & 1;            // 0..1 : 32-col strip
    const int m0   = blockIdx.y * 128;
    const int n0   = blockIdx.x * 64;

    v8f acc[2][2];
    for (int i = 0; i < 2; i++)
        for (int j = 0; j < 2; j++) acc[i][j] = zero8();

    const int bk = tid >> 3;            // 0..31 B k row
    const int bn = (tid & 7) * 8;       // 0..56 B col group

    const int r  = lane & 15;
    const int hf = lane >> 4;

    const unsigned lds_a = (unsigned)(size_t)(&As32[0]);

    for (int k0 = 0; k0 < K; k0 += 32) {
        __syncthreads();

        // ---- issue TDM load of the A tile (wave 0 only) ----
        if (wv == 0) {
            unsigned long long ga =
                (unsigned long long)(size_t)(A + (size_t)m0 * K + k0);
            u32x4 g0;
            g0[0] = 1u;                                   // count=1 descriptor
            g0[1] = lds_a;                                // lds_addr (bytes)
            g0[2] = (unsigned)(ga & 0xFFFFFFFFu);         // global_addr[31:0]
            g0[3] = (unsigned)((ga >> 32) & 0x1FFFFFFu)   // global_addr[56:32]
                  | (2u << 30);                           // type=2 (image)
            i32x8 g1;
            g1[0] = (int)((2u << 16)      // data_size = 4B
                        | (1u << 20)      // pad_enable
                        | (4u << 22)      // pad_interval: 32 DWORDs
                        | (7u << 25));    // pad_amount: 8 DWORDs
            g1[1] = (int)(((unsigned)K & 0xFFFFu) << 16);         // tdim0 lo
            g1[2] = (int)((((unsigned)K >> 16) & 0xFFFFu)
                        | (((unsigned)M & 0xFFFFu) << 16));       // tdim0 hi | tdim1 lo
            g1[3] = (int)((((unsigned)M >> 16) & 0xFFFFu)
                        | (32u << 16));                           // tdim1 hi | tile_dim0=32
            g1[4] = 128;                                          // tile_dim1=128
            g1[5] = (int)(unsigned)K;                             // tensor_dim0_stride
            g1[6] = 0;
            g1[7] = 0;
            i32x4 gz4 = {0, 0, 0, 0};
            i32x8 gz8 = {0, 0, 0, 0, 0, 0, 0, 0};
            __builtin_amdgcn_tensor_load_to_lds(g0, g1, gz4, gz4, gz8, 0);
        }

        // ---- stage B tile manually (32x64 fp32 -> transposed f16 [n][k]) ----
        const float* bp = Bw + (size_t)(k0 + bk) * N + n0 + bn;
        __builtin_prefetch(bp + (size_t)32 * N, 0, 0);
        {
            float4 f0 = *(const float4*)(bp + 0);
            float4 f1 = *(const float4*)(bp + 4);
            Bs[(bn + 0) * 40 + bk] = (_Float16)f0.x;
            Bs[(bn + 1) * 40 + bk] = (_Float16)f0.y;
            Bs[(bn + 2) * 40 + bk] = (_Float16)f0.z;
            Bs[(bn + 3) * 40 + bk] = (_Float16)f0.w;
            Bs[(bn + 4) * 40 + bk] = (_Float16)f1.x;
            Bs[(bn + 5) * 40 + bk] = (_Float16)f1.y;
            Bs[(bn + 6) * 40 + bk] = (_Float16)f1.z;
            Bs[(bn + 7) * 40 + bk] = (_Float16)f1.w;
        }

        if (wv == 0) __builtin_amdgcn_s_wait_tensorcnt(0);
        __syncthreads();

        // ---- fragments: lane r = row/col %16, half hf selects K groups ----
        FragH a[2], b[2];
        #pragma unroll
        for (int s = 0; s < 2; s++) {
            const float* pa = &As32[(wm * 32 + s * 16 + r) * 40 + hf * 8];
            load8_h(&a[s].h[0], pa,      1.0f);   // K = hf*8 .. +7
            load8_h(&a[s].h[8], pa + 16, 1.0f);   // K = 16+hf*8 .. +7
            const _Float16* pb = &Bs[(wn * 32 + s * 16 + r) * 40 + hf * 8];
            b[s].q[0] = *(const uint4*)pb;
            b[s].q[1] = *(const uint4*)(pb + 16);
        }
        #pragma unroll
        for (int i = 0; i < 2; i++)
            #pragma unroll
            for (int j = 0; j < 2; j++)
                acc[i][j] = wmma16(a[i].v, b[j].v, acc[i][j]);
    }

    // epilogue: VGPR e -> row (hf*8+e) in 16-tile, col r; fused bias (+ReLU)
    #pragma unroll
    for (int i = 0; i < 2; i++) {
        #pragma unroll
        for (int j = 0; j < 2; j++) {
            const int col = n0 + wn * 32 + j * 16 + r;
            const float bz = bias[col];
            #pragma unroll
            for (int e = 0; e < 8; e++) {
                int row = m0 + wm * 32 + i * 16 + hf * 8 + e;
                float v = acc[i][j][e] + bz;
                if (relu) v = fmaxf(v, 0.0f);
                C[(size_t)row * N + col] = v;
            }
        }
    }
}

// ---------------------------------------------------------------------------
// Banded attention: one workgroup per (b, h, chunk c, 32-query sub-block)
// keys j in [c*W - W, c*W + 2W), mask |j - i| <= W and 0 <= j < S
// ---------------------------------------------------------------------------
__global__ __launch_bounds__(256)
void attn_kernel(const float* __restrict__ xin, const float* __restrict__ qb,
                 const float* __restrict__ kb, const float* __restrict__ vb,
                 float* __restrict__ xa)
{
    __shared__ _Float16 sc[32 * 392];   // scores then probs, stride 392 (16B ok)

    int g = blockIdx.x;
    const int sub = g & 3;        g >>= 2;
    const int cc  = g & (NC_ - 1); g >>= 5;
    const int hh  = g % H_;
    const int bb  = g / H_;

    const int tid  = threadIdx.x;
    const int lane = tid & 31;
    const int wv   = tid >> 5;
    const int r    = lane & 15;
    const int hf   = lane >> 4;

    const size_t headoff = (size_t)hh * DH_;
    const int qrow0 = cc * W_ + sub * 32;   // first global query row of this WG

    // ---- stage 1: scores = (q/sqrt(DH)) . k  -> masked f16 in LDS ----
    {
        const int qsub = wv & 1;            // which 16-query tile
        const int kblk = (wv >> 1) * 96;    // 6 key tiles of 16 per wave

        FragH aq[2];                        // q fragment, K = d (64 -> 2 frags)
        {
            const int i = qrow0 + qsub * 16 + r;
            const float* qp = qb + ((size_t)(bb * S_ + i)) * D_ + headoff;
            const float scale = 0.125f;     // 1/sqrt(64)
            #pragma unroll
            for (int f = 0; f < 2; f++) {
                load8_h(&aq[f].h[0], qp + f * 32 + hf * 8,      scale);
                load8_h(&aq[f].h[8], qp + f * 32 + 16 + hf * 8, scale);
            }
        }
        for (int kt = 0; kt < 6; kt++) {
            const int ks = kblk + kt * 16;
            const int j  = cc * W_ + ks + r - W_;   // this lane's key row
            FragH bk2[2];
            if (j >= 0 && j < S_) {
                const float* kp = kb + ((size_t)(bb * S_ + j)) * D_ + headoff;
                #pragma unroll
                for (int f = 0; f < 2; f++) {
                    load8_h(&bk2[f].h[0], kp + f * 32 + hf * 8,      1.0f);
                    load8_h(&bk2[f].h[8], kp + f * 32 + 16 + hf * 8, 1.0f);
                }
            } else {
                #pragma unroll
                for (int f = 0; f < 2; f++)
                    #pragma unroll
                    for (int e = 0; e < 16; e++) bk2[f].h[e] = (_Float16)0.0f;
            }
            v8f accs = zero8();
            accs = wmma16(aq[0].v, bk2[0].v, accs);
            accs = wmma16(aq[1].v, bk2[1].v, accs);
            // mask + store
            #pragma unroll
            for (int e = 0; e < 8; e++) {
                int qi  = qsub * 16 + hf * 8 + e;   // 0..31 in-WG query
                int ki  = ks + r;                   // 0..383
                int i_g = qrow0 + qi;
                int j_g = cc * W_ + ki - W_;
                bool valid = (j_g >= 0) & (j_g < S_) &
                             (j_g - i_g <= W_) & (i_g - j_g <= W_);
                sc[qi * 392 + ki] = (_Float16)(valid ? accs[e] : -65504.0f);
            }
        }
    }
    __syncthreads();

    // ---- stage 2: row softmax (fp32 math over f16 scores) ----
    if (tid < 32) {
        _Float16* row = &sc[tid * 392];
        float m = -1e30f;
        for (int k = 0; k < 384; k++) m = fmaxf(m, (float)row[k]);
        float ssum = 0.0f;
        for (int k = 0; k < 384; k++) {
            float v = __expf((float)row[k] - m);
            ssum += v;
            row[k] = (_Float16)v;
        }
        float inv = 1.0f / ssum;
        for (int k = 0; k < 384; k++)
            row[k] = (_Float16)((float)row[k] * inv);
    }
    __syncthreads();

    // ---- stage 3: out = probs @ v, fused with residual add ----
    {
        const int qsub = wv >> 2;       // 0..1
        const int dsub = wv & 3;        // 0..3 (16 cols each of DH=64)
        v8f acc = zero8();
        for (int kt = 0; kt < 12; kt++) {       // K = 384 keys, 32 per step
            FragH ap, bv2;
            const _Float16* pp = &sc[(qsub * 16 + r) * 392 + kt * 32 + hf * 8];
            ap.q[0] = *(const uint4*)pp;
            ap.q[1] = *(const uint4*)(pp + 16);
            const int d = dsub * 16 + r;
            #pragma unroll
            for (int e = 0; e < 16; e++) {
                int jl = kt * 32 + (e < 8 ? 0 : 16) + hf * 8 + (e & 7);
                int j  = cc * W_ + jl - W_;
                float v = (j >= 0 && j < S_)
                        ? vb[((size_t)(bb * S_ + j)) * D_ + headoff + d]
                        : 0.0f;
                bv2.h[e] = (_Float16)v;
            }
            acc = wmma16(ap.v, bv2.v, acc);
        }
        #pragma unroll
        for (int e = 0; e < 8; e++) {
            int qi  = qsub * 16 + hf * 8 + e;
            int i_g = qrow0 + qi;
            int d   = dsub * 16 + r;
            size_t idx = ((size_t)(bb * S_ + i_g)) * D_ + headoff + d;
            xa[idx] = xin[idx] + acc[e];
        }
    }
}

// ---------------------------------------------------------------------------
// Fused residual add + LayerNorm over D=768; one block per row
// ---------------------------------------------------------------------------
__global__ __launch_bounds__(256)
void ln_kernel(const float* __restrict__ a, const float* __restrict__ b,
               const float* __restrict__ g, const float* __restrict__ beta,
               float* __restrict__ out)
{
    __shared__ float rs[256], rs2[256];
    const int row = blockIdx.x;
    const int tid = threadIdx.x;
    const float* ap = a + (size_t)row * D_;
    float x[3];
    float s = 0.f, s2 = 0.f;
    #pragma unroll
    for (int i = 0; i < 3; i++) {
        int c = tid + i * 256;
        float v = ap[c];
        if (b) v += b[(size_t)row * D_ + c];
        x[i] = v; s += v; s2 += v * v;
    }
    rs[tid] = s; rs2[tid] = s2;
    __syncthreads();
    for (int st = 128; st > 0; st >>= 1) {
        if (tid < st) { rs[tid] += rs[tid + st]; rs2[tid] += rs2[tid + st]; }
        __syncthreads();
    }
    float mean = rs[0] * (1.0f / D_);
    float var  = rs2[0] * (1.0f / D_) - mean * mean;
    float inv  = rsqrtf(var + 1e-6f);
    #pragma unroll
    for (int i = 0; i < 3; i++) {
        int c = tid + i * 256;
        out[(size_t)row * D_ + c] = (x[i] - mean) * inv * g[c] + beta[c];
    }
}

// ---------------------------------------------------------------------------
extern "C" void kernel_launch(void* const* d_in, const int* in_sizes, int n_in,
                              void* d_out, int out_size, void* d_ws, size_t ws_size,
                              hipStream_t stream) {
    const float* src  = (const float*)d_in[0];
    const float* Wq   = (const float*)d_in[1];
    const float* bq   = (const float*)d_in[2];
    const float* Wk   = (const float*)d_in[3];
    const float* bk   = (const float*)d_in[4];
    const float* Wv   = (const float*)d_in[5];
    const float* bv   = (const float*)d_in[6];
    const float* W1   = (const float*)d_in[7];
    const float* b1   = (const float*)d_in[8];
    const float* W2   = (const float*)d_in[9];
    const float* b2   = (const float*)d_in[10];
    const float* ln2g = (const float*)d_in[11];
    const float* ln2b = (const float*)d_in[12];
    const float* lnfg = (const float*)d_in[13];
    const float* lnfb = (const float*)d_in[14];
    float* out = (float*)d_out;
    float* ws  = (float*)d_ws;

    const size_t NT = (size_t)B_ * S_ * D_;     // 6.29M floats
    float* x  = ws;
    float* qb = ws + 1 * NT;
    float* kb = ws + 2 * NT;
    float* vb = ws + 3 * NT;
    float* xa = ws + 4 * NT;
    float* t2 = ws + 5 * NT;
    float* t1 = ws + 6 * NT;                    // B*S*DFF floats

    dim3 blk(256);
    dim3 gP (D_   / 64, MROWS / 128);   // 12 x 64
    dim3 gF1(DFF_ / 64, MROWS / 128);   // 48 x 64
    dim3 gA (B_ * H_ * NC_ * 4);        // 3072
    dim3 gLN(MROWS);                    // 8192

    const float* xin = src;
    for (int l = 0; l < L_; l++) {
        gemm_kernel<<<gP,  blk, 0, stream>>>(xin, Wq + (size_t)l * D_ * D_,  bq + l * D_,   qb, MROWS, D_,   D_,   0);
        gemm_kernel<<<gP,  blk, 0, stream>>>(xin, Wk + (size_t)l * D_ * D_,  bk + l * D_,   kb, MROWS, D_,   D_,   0);
        gemm_kernel<<<gP,  blk, 0, stream>>>(xin, Wv + (size_t)l * D_ * D_,  bv + l * D_,   vb, MROWS, D_,   D_,   0);
        attn_kernel<<<gA,  blk, 0, stream>>>(xin, qb, kb, vb, xa);
        gemm_kernel<<<gF1, blk, 0, stream>>>(xa,  W1 + (size_t)l * D_ * DFF_, b1 + l * DFF_, t1, MROWS, D_,   DFF_, 1);
        gemm_kernel<<<gP,  blk, 0, stream>>>(t1,  W2 + (size_t)l * DFF_ * D_, b2 + l * D_,   t2, MROWS, DFF_, D_,   0);
        ln_kernel<<<gLN,   blk, 0, stream>>>(xa, t2, ln2g + l * D_, ln2b + l * D_, x);
        xin = x;
    }
    ln_kernel<<<gLN, blk, 0, stream>>>(x, nullptr, lnfg, lnfb, out);
}
